// DPAP2PNet_78932908966023
// MI455X (gfx1250) — matrix operand order repaired and checked by reading, not compiled, not run on hardware
//
#include <hip/hip_runtime.h>

typedef __attribute__((ext_vector_type(16))) __bf16 v16bf;
typedef __attribute__((ext_vector_type(8)))  float  v8f;
typedef __bf16 bf16;

// ---------- helpers ----------
__device__ __forceinline__ v16bf ld16(const bf16* p0, const bf16* p1) {
  union { v16bf v; uint4 q[2]; } u;
  u.q[0] = *(const uint4*)p0;
  u.q[1] = *(const uint4*)p1;
  return u.v;
}

__device__ __forceinline__ v8f wmma_bf16(v16bf a, v16bf b, v8f c) {
  // D = A(16x32 bf16) * B(32x16 bf16) + C(16x16 f32)
  return __builtin_amdgcn_wmma_f32_16x16x32_bf16(false, a, false, b, (short)0, c, false, false);
}

// Software-pipelined GEMM core: 16 N-tiles per K-chunk, B prefetched one
// 4-tile group ahead, A prefetched one chunk ahead. bpack chunk stride =
// 16 nt * 512 elems = 8192 elems; group stride = 2048 elems.
template <int NCH, typename AF>
__device__ __forceinline__ void gemm_core(AF aptr, const bf16* __restrict__ bpack,
                                          int lane, v8f* acc) {
  const bf16* ap = aptr(0);
  v16bf a = ld16(ap, ap + 16);
  const bf16* b0 = bpack + lane * 16;
  v16bf bb0 = ld16(b0, b0 + 8);
  v16bf bb1 = ld16(b0 + 512, b0 + 520);
  v16bf bb2 = ld16(b0 + 1024, b0 + 1032);
  v16bf bb3 = ld16(b0 + 1536, b0 + 1544);
  for (int c = 0; c < NCH; ++c) {
    const bf16* bc = bpack + (size_t)c * 8192 + lane * 16;
    // WGP-scope prefetch (locality 3 -> scope 0): pull next-next chunk of the
    // weight pack into all cache levels while ~32 WMMAs away.
    if (c + 2 < NCH) __builtin_prefetch(bc + 16384, 0, 3);  // global_prefetch_b8
    v16bf an = a;
#pragma unroll
    for (int g = 0; g < 4; ++g) {
      const bf16* pf;
      if (g < 3) pf = bc + (g + 1) * 2048;
      else pf = (c + 1 < NCH) ? bpack + (size_t)(c + 1) * 8192 + lane * 16 : bc;
      v16bf n0 = ld16(pf, pf + 8);
      v16bf n1 = ld16(pf + 512, pf + 520);
      v16bf n2 = ld16(pf + 1024, pf + 1032);
      v16bf n3 = ld16(pf + 1536, pf + 1544);
      if (g == 3) {
        const bf16* apn = aptr(c + 1 < NCH ? c + 1 : c);
        an = ld16(apn, apn + 16);
      }
      acc[4 * g + 0] = wmma_bf16(a, bb0, acc[4 * g + 0]);
      acc[4 * g + 1] = wmma_bf16(a, bb1, acc[4 * g + 1]);
      acc[4 * g + 2] = wmma_bf16(a, bb2, acc[4 * g + 2]);
      acc[4 * g + 3] = wmma_bf16(a, bb3, acc[4 * g + 3]);
      bb0 = n0; bb1 = n1; bb2 = n2; bb3 = n3;
    }
    a = an;
  }
}

// ---------- 1) offsets MLP + proposals ----------
__global__ __launch_bounds__(256) void k_prop(const float* __restrict__ o,
                                              const float* __restrict__ dw1,
                                              const float* __restrict__ db1,
                                              const float* __restrict__ dw2,
                                              const float* __restrict__ db2,
                                              float* __restrict__ prop_ws,
                                              float* __restrict__ out_prop) {
  __shared__ float smem[16 * 768];  // 48KB, reused for hidden[16][256]
  int blk = blockIdx.x;             // b*16 + i
  int b = blk >> 4, i = blk & 15;
  int t = threadIdx.x;
  const float* obase = o + ((size_t)(b * 16 + i) * 16) * 768;
  for (int idx = t; idx < 16 * 768; idx += 256) smem[idx] = obase[idx];
  __syncthreads();
  float acc[16];
#pragma unroll
  for (int p = 0; p < 16; ++p) acc[p] = db1[t];
  for (int k = 0; k < 768; ++k) {
    float w = dw1[k * 256 + t];
#pragma unroll
    for (int p = 0; p < 16; ++p) acc[p] = fmaf(smem[p * 768 + k], w, acc[p]);
  }
  __syncthreads();  // done reading o
#pragma unroll
  for (int p = 0; p < 16; ++p) smem[p * 256 + t] = fmaxf(acc[p], 0.f);
  __syncthreads();
  if (t < 128) {
    int p = t >> 3, jo = t & 7;  // p = j index, jo = (a*4 + c*2 + e)
    float s = db2[jo];
    for (int c = 0; c < 256; ++c) s = fmaf(smem[p * 256 + c], dw2[c * 8 + jo], s);
    int a = jo >> 2, cq = (jo >> 1) & 1, e = jo & 1;
    int y = 2 * i + a, x = 2 * p + cq;
    float anchor = (e == 0) ? (x * 16.f + 8.f) : (y * 16.f + 8.f);
    float v = anchor + s;
    size_t oi = (((size_t)b * 32 + y) * 32 + x) * 2 + e;
    prop_ws[oi] = v;
    out_prop[oi] = v;
  }
}

// ---------- 2) NCHW f32 -> channels-last bf16 (optionally padded) ----------
__global__ __launch_bounds__(256) void k_tocl(const float* __restrict__ src, bf16* __restrict__ dst,
                                              int H, int W, size_t bStride, size_t rowStride,
                                              size_t baseOff) {
  __shared__ float tile[32 * 129];
  int blk = blockIdx.x;
  int cg = blk & 7;
  int y = (blk >> 3) % H;
  int b = blk / (8 * H);
  int c0 = cg * 32;
  int t = threadIdx.x;
  const float* s = src + ((size_t)(b * 256 + c0) * H + y) * W;
  for (int idx = t; idx < 32 * W; idx += 256) {
    int c = idx / W, x = idx - c * W;
    tile[c * 129 + x] = s[(size_t)c * H * W + x];
  }
  __syncthreads();
  bf16* d = dst + baseOff + (size_t)b * bStride + (size_t)y * rowStride + c0;
  for (int idx = t; idx < 32 * W; idx += 256) {
    int x = idx >> 5, c = idx & 31;
    d[(size_t)x * 256 + c] = (bf16)tile[c * 129 + x];
  }
}

// zero the 1-wide ring of a padded [8][P][P][C] bf16 buffer
__global__ __launch_bounds__(256) void k_zero_ring(bf16* __restrict__ base, int P, int C) {
  int npos = 2 * P + 2 * (P - 2);
  int blk = blockIdx.x;
  int b = blk / npos, pos = blk - b * npos;
  int y, x;
  if (pos < P)               { y = 0;     x = pos; }
  else if (pos < 2 * P)      { y = P - 1; x = pos - P; }
  else if (pos < 2 * P + (P - 2)) { y = pos - 2 * P + 1; x = 0; }
  else                       { y = pos - (2 * P + (P - 2)) + 1; x = P - 1; }
  bf16* p = base + (((size_t)b * P + y) * P + x) * C;
  for (int i = threadIdx.x; i < C; i += 256) p[i] = (bf16)0.f;
}

// ---------- 3) grid-sample 4 levels into padded channels-last roi ----------
__global__ __launch_bounds__(256) void k_sample(const float* __restrict__ prop,
                                                const bf16* __restrict__ f0, const bf16* __restrict__ f1,
                                                const bf16* __restrict__ f2, const bf16* __restrict__ f3,
                                                bf16* __restrict__ roipad) {
  int blk = blockIdx.x;
  int b = blk >> 10, pt = blk & 1023;
  int py = pt >> 5, px = pt & 31;
  int c = threadIdx.x;
  float fx = prop[(size_t)blk * 2 + 0];
  float fy = prop[(size_t)blk * 2 + 1];
  const bf16* fcl[4] = {f0, f1, f2, f3};
  bf16* dst = roipad + (((size_t)(b * 34 + py + 1) * 34) + (px + 1)) * 1024 + c;
#pragma unroll
  for (int l = 0; l < 4; ++l) {
    const int fw = 128 >> l;            // 128,64,32,16 (square)
    const float s = (float)(4 << l);    // 4,8,16,32
    float xs = fx * (float)(fw - 1) / (s * (float)fw);
    float ys = fy * (float)(fw - 1) / (s * (float)fw);
    float x0f = floorf(xs), y0f = floorf(ys);
    float wx1 = xs - x0f, wy1 = ys - y0f;
    int x0 = (int)x0f, y0 = (int)y0f;
    float v = 0.f;
#pragma unroll
    for (int dy = 0; dy < 2; ++dy)
#pragma unroll
      for (int dx = 0; dx < 2; ++dx) {
        int xi = x0 + dx, yi = y0 + dy;
        if (xi >= 0 && xi < fw && yi >= 0 && yi < fw) {
          float w = (dx ? wx1 : 1.f - wx1) * (dy ? wy1 : 1.f - wy1);
          float fv = (float)fcl[l][(((size_t)(b * fw + yi) * fw) + xi) * 256 + c];
          v = fmaf(fv, w, v);
        }
      }
    dst[l * 256] = (bf16)v;
  }
}

// ---------- 4) weight repack into per-lane WMMA B layout ----------
// dst id = ((c*NT+nt)*32+lane)*16+e ; k = c*32 + (lane>=16?8:0) + (e&7) + (e>=8?16:0) ; n = nt*16 + (lane&15)
__global__ __launch_bounds__(256) void k_pack(const float* __restrict__ src, bf16* __restrict__ dst,
                                              int K, int N, int mode, int IC, int Nsrc) {
  size_t id = (size_t)blockIdx.x * 256 + threadIdx.x;
  if (id >= (size_t)K * N) return;
  int e = id & 15;
  int lane = (id >> 4) & 31;
  size_t rest = id >> 9;
  int NT = N >> 4;
  int nt = (int)(rest % NT);
  int c = (int)(rest / NT);
  int k = c * 32 + ((lane >= 16) ? 8 : 0) + (e & 7) + ((e >= 8) ? 16 : 0);
  int n = nt * 16 + (lane & 15);
  float v = 0.f;
  if (k < K && n < Nsrc) {
    if (mode == 0) {
      int kpos = k / IC, ic = k - kpos * IC;
      int ky = kpos / 3, kx = kpos - ky * 3;
      v = src[(((size_t)n * IC + ic) * 3 + ky) * 3 + kx];
    } else {
      v = src[(size_t)k * Nsrc + n];
    }
  }
  dst[id] = (bf16)v;
}

// fold BN(eval) + conv bias into per-channel scale/bias
__global__ void k_bnfold(const float* g, const float* bb, const float* mn, const float* vr,
                         const float* mb1, float* sc, float* bi) {
  int c = threadIdx.x;
  float s = g[c] * rsqrtf(vr[c] + 1e-5f);
  sc[c] = s;
  bi[c] = mb1[c] * s + bb[c] - mn[c] * s;
}

// ---------- 5) ROI 3x3 conv GEMM: K=9216, + conv_b -> bf16 CL ----------
__global__ __launch_bounds__(32) void k_gemm_roi(const bf16* __restrict__ roipad,
                                                 const bf16* __restrict__ bpack,
                                                 const float* __restrict__ bias,
                                                 bf16* __restrict__ out) {
  int wg = blockIdx.x;  // 8*32*2
  int xt = wg & 1, y = (wg >> 1) & 31, b = wg >> 6;
  int lane = threadIdx.x;
  int m = lane & 15;
  int kbase = (lane < 16) ? 0 : 8;
  int x = xt * 16 + m;
  v8f acc[16];
#pragma unroll
  for (int nt = 0; nt < 16; ++nt)
#pragma unroll
    for (int r = 0; r < 8; ++r) acc[nt][r] = 0.f;
  auto af = [&](int c) -> const bf16* {
    int ky = c / 96, kx = (c / 32) % 3, ic0 = (c & 31) * 32;
    return roipad + (((size_t)(b * 34 + y + ky) * 34 + (x + kx)) * 1024 + ic0 + kbase);
  };
  gemm_core<288>(af, bpack, lane, acc);
  int n0 = lane & 15;
  int mofs = (lane < 16) ? 0 : 8;
#pragma unroll
  for (int nt = 0; nt < 16; ++nt) {
    int ch = nt * 16 + n0;
    float bv = bias[ch];
#pragma unroll
    for (int r = 0; r < 8; ++r) {
      int xx = xt * 16 + r + mofs;
      out[(((size_t)(b * 32 + y) * 32 + xx) * 256) + ch] = (bf16)(acc[nt][r] + bv);
    }
  }
}

// ---------- 6) mask 3x3 conv GEMM (K=2304) fused BN+ReLU+1x1 dot ----------
__global__ __launch_bounds__(32) void k_gemm_mask(const bf16* __restrict__ mpad,
                                                  const bf16* __restrict__ bpack,
                                                  const float* __restrict__ sc,
                                                  const float* __restrict__ bi,
                                                  const float* __restrict__ w2,
                                                  const float* __restrict__ mb2p,
                                                  float* __restrict__ mdot) {
  int wg = blockIdx.x;  // 8*128*8
  int xt = wg & 7, y = (wg >> 3) & 127, b = wg >> 10;
  int lane = threadIdx.x;
  int m = lane & 15;
  int kbase = (lane < 16) ? 0 : 8;
  int x = xt * 16 + m;
  v8f acc[16];
#pragma unroll
  for (int nt = 0; nt < 16; ++nt)
#pragma unroll
    for (int r = 0; r < 8; ++r) acc[nt][r] = 0.f;
  auto af = [&](int c) -> const bf16* {
    int ky = c / 24, kx = (c / 8) % 3, ic0 = (c & 7) * 32;
    return mpad + (((size_t)(b * 130 + y + ky) * 130 + (x + kx)) * 256 + ic0 + kbase);
  };
  gemm_core<72>(af, bpack, lane, acc);
  // epilogue: per-channel BN+ReLU then dot with mw2
  int n0 = lane & 15;
  float part[8];
#pragma unroll
  for (int r = 0; r < 8; ++r) part[r] = 0.f;
#pragma unroll
  for (int nt = 0; nt < 16; ++nt) {
    int ch = nt * 16 + n0;
    float s = sc[ch], bb = bi[ch], w = w2[ch];
#pragma unroll
    for (int r = 0; r < 8; ++r) {
      float vv = fmaxf(fmaf(acc[nt][r], s, bb), 0.f);
      part[r] = fmaf(vv, w, part[r]);
    }
  }
  __shared__ float red[32][8];
#pragma unroll
  for (int r = 0; r < 8; ++r) red[lane][r] = part[r];
  __syncthreads();
  if (lane < 16) {
    int mm = lane;
    float ssum = 0.f;
    if (mm < 8) {
      for (int l = 0; l < 16; ++l) ssum += red[l][mm];
    } else {
      for (int l = 0; l < 16; ++l) ssum += red[16 + l][mm - 8];
    }
    mdot[((size_t)(b * 128 + y) * 128) + xt * 16 + mm] = ssum + mb2p[0];
  }
}

// ---------- 7) classifier FC1 (relu) and FC2 ----------
__global__ __launch_bounds__(32) void k_fc1(const bf16* __restrict__ xin, const bf16* __restrict__ bpack,
                                            const float* __restrict__ bias, bf16* __restrict__ h) {
  int tile = blockIdx.x;  // 512
  int row0 = tile * 16;
  int lane = threadIdx.x;
  int m = lane & 15;
  int kbase = (lane < 16) ? 0 : 8;
  v8f acc[16];
#pragma unroll
  for (int nt = 0; nt < 16; ++nt)
#pragma unroll
    for (int r = 0; r < 8; ++r) acc[nt][r] = 0.f;
  auto af = [&](int c) -> const bf16* {
    return xin + ((size_t)(row0 + m) * 256 + c * 32 + kbase);
  };
  gemm_core<8>(af, bpack, lane, acc);
  int n0 = lane & 15;
  int mofs = (lane < 16) ? 0 : 8;
#pragma unroll
  for (int nt = 0; nt < 16; ++nt) {
    int ch = nt * 16 + n0;
    float bv = bias[ch];
#pragma unroll
    for (int r = 0; r < 8; ++r)
      h[(size_t)(row0 + r + mofs) * 256 + ch] = (bf16)fmaxf(acc[nt][r] + bv, 0.f);
  }
}

__global__ __launch_bounds__(32) void k_fc2(const bf16* __restrict__ h, const bf16* __restrict__ bpack,
                                            const float* __restrict__ bias, float* __restrict__ logits) {
  int tile = blockIdx.x;  // 512
  int row0 = tile * 16;
  int lane = threadIdx.x;
  int m = lane & 15;
  int kbase = (lane < 16) ? 0 : 8;
  v8f acc;
#pragma unroll
  for (int r = 0; r < 8; ++r) acc[r] = 0.f;
  // pipelined: prefetch next chunk's A and B before each WMMA
  const bf16* ar = h + ((size_t)(row0 + m) * 256 + kbase);
  const bf16* bp = bpack + lane * 16;
  v16bf a = ld16(ar, ar + 16);
  v16bf bm = ld16(bp, bp + 8);
#pragma unroll
  for (int c = 0; c < 8; ++c) {
    v16bf an = a, bn = bm;
    if (c + 1 < 8) {
      const bf16* arn = ar + (c + 1) * 32;
      const bf16* bpn = bp + (size_t)(c + 1) * 512;
      an = ld16(arn, arn + 16);
      bn = ld16(bpn, bpn + 8);
    }
    acc = wmma_bf16(a, bm, acc);
    a = an; bm = bn;
  }
  int n0 = lane & 15;
  int mofs = (lane < 16) ? 0 : 8;
  if (n0 < 6) {
    float bv = bias[n0];
#pragma unroll
    for (int r = 0; r < 8; ++r)
      logits[(size_t)(row0 + r + mofs) * 6 + n0] = acc[r] + bv;
  }
}

// ---------- 8) bilinear upsample (130x130 virtual, border=mb2) -> 512x512 ----------
__global__ __launch_bounds__(256) void k_upsample(const float* __restrict__ mdot,
                                                  const float* __restrict__ mb2p,
                                                  float* __restrict__ out) {
  int id = blockIdx.x * 256 + threadIdx.x;  // 8*512*512
  int ox = id & 511, oy = (id >> 9) & 511, b = id >> 18;
  float mb2 = mb2p[0];
  const float s = 129.0f / 511.0f;
  float ysf = oy * s, xsf = ox * s;
  int y0 = (int)floorf(ysf), x0 = (int)floorf(xsf);
  float wy = ysf - y0, wx = xsf - x0;
  int y1 = (y0 + 1 < 129) ? y0 + 1 : 129;
  int x1 = (x0 + 1 < 129) ? x0 + 1 : 129;
  auto fetch = [&](int yy, int xx) -> float {
    if (yy >= 1 && yy <= 128 && xx >= 1 && xx <= 128)
      return mdot[((size_t)b * 128 + (yy - 1)) * 128 + (xx - 1)];
    return mb2;
  };
  float top = fetch(y0, x0) * (1.f - wx) + fetch(y0, x1) * wx;
  float bot = fetch(y1, x0) * (1.f - wx) + fetch(y1, x1) * wx;
  out[id] = top * (1.f - wy) + bot * wy;
}

// ---------- launcher ----------
extern "C" void kernel_launch(void* const* d_in, const int* in_sizes, int n_in,
                              void* d_out, int out_size, void* d_ws, size_t ws_size,
                              hipStream_t stream) {
  (void)in_sizes; (void)n_in; (void)out_size; (void)ws_size;
  const float* feat0 = (const float*)d_in[1];
  const float* feat1 = (const float*)d_in[2];
  const float* feat2 = (const float*)d_in[3];
  const float* feat3 = (const float*)d_in[4];
  const float* mask_feat = (const float*)d_in[5];
  const float* o    = (const float*)d_in[6];
  const float* dw1  = (const float*)d_in[7];
  const float* db1  = (const float*)d_in[8];
  const float* dw2  = (const float*)d_in[9];
  const float* db2  = (const float*)d_in[10];
  const float* cw1  = (const float*)d_in[11];
  const float* cb1  = (const float*)d_in[12];
  const float* cw2  = (const float*)d_in[13];
  const float* cb2  = (const float*)d_in[14];
  const float* conv_w = (const float*)d_in[15];
  const float* conv_b = (const float*)d_in[16];
  const float* mw1  = (const float*)d_in[17];
  const float* mb1  = (const float*)d_in[18];
  const float* bn_g = (const float*)d_in[19];
  const float* bn_b = (const float*)d_in[20];
  const float* bn_m = (const float*)d_in[21];
  const float* bn_v = (const float*)d_in[22];
  const float* mw2  = (const float*)d_in[23];
  const float* mb2  = (const float*)d_in[24];
  float* out = (float*)d_out;

  char* ws = (char*)d_ws;
  size_t off = 0;
  auto alloc = [&](size_t bytes) { size_t o2 = off; off += (bytes + 255) & ~(size_t)255; return o2; };
  size_t PROP  = alloc((size_t)8 * 1024 * 2 * 4);
  size_t F0CL  = alloc((size_t)8 * 128 * 128 * 256 * 2);
  size_t F1CL  = alloc((size_t)8 * 64 * 64 * 256 * 2);
  size_t F2CL  = alloc((size_t)8 * 32 * 32 * 256 * 2);
  size_t F3CL  = alloc((size_t)8 * 16 * 16 * 256 * 2);
  size_t MPAD  = F0CL;  // alias: mask pad (69.2MB) reuses F0CL+F1CL (83.9MB), dead after sampling
  size_t ROIPAD = alloc((size_t)8 * 34 * 34 * 1024 * 2);
  size_t WPC   = alloc((size_t)9216 * 256 * 2);
  size_t WPM   = alloc((size_t)2304 * 256 * 2);
  size_t WP1   = alloc((size_t)256 * 256 * 2);
  size_t WP2   = alloc((size_t)256 * 16 * 2);
  size_t ROICL = alloc((size_t)8192 * 256 * 2);
  size_t HBUF  = alloc((size_t)8192 * 256 * 2);
  size_t MDOT  = alloc((size_t)8 * 128 * 128 * 4);
  size_t BNSC  = alloc(256 * 4);
  size_t BNBI  = alloc(256 * 4);

  float* prop_ws = (float*)(ws + PROP);
  bf16* f0cl = (bf16*)(ws + F0CL);
  bf16* f1cl = (bf16*)(ws + F1CL);
  bf16* f2cl = (bf16*)(ws + F2CL);
  bf16* f3cl = (bf16*)(ws + F3CL);
  bf16* mpad = (bf16*)(ws + MPAD);
  bf16* roipad = (bf16*)(ws + ROIPAD);
  bf16* wpc = (bf16*)(ws + WPC);
  bf16* wpm = (bf16*)(ws + WPM);
  bf16* wp1 = (bf16*)(ws + WP1);
  bf16* wp2 = (bf16*)(ws + WP2);
  bf16* roicl = (bf16*)(ws + ROICL);
  bf16* hbuf = (bf16*)(ws + HBUF);
  float* mdot = (float*)(ws + MDOT);
  float* bnsc = (float*)(ws + BNSC);
  float* bnbi = (float*)(ws + BNBI);

  // 1) proposals (out[0..16384))
  k_prop<<<128, 256, 0, stream>>>(o, dw1, db1, dw2, db2, prop_ws, out);

  // 2) features -> channels-last bf16
  k_tocl<<<8 * 128 * 8, 256, 0, stream>>>(feat0, f0cl, 128, 128, (size_t)128 * 128 * 256, (size_t)128 * 256, 0);
  k_tocl<<<8 * 64 * 8, 256, 0, stream>>>(feat1, f1cl, 64, 64, (size_t)64 * 64 * 256, (size_t)64 * 256, 0);
  k_tocl<<<8 * 32 * 8, 256, 0, stream>>>(feat2, f2cl, 32, 32, (size_t)32 * 32 * 256, (size_t)32 * 256, 0);
  k_tocl<<<8 * 16 * 8, 256, 0, stream>>>(feat3, f3cl, 16, 16, (size_t)16 * 16 * 256, (size_t)16 * 256, 0);
  k_zero_ring<<<8 * 132, 256, 0, stream>>>(roipad, 34, 1024);

  // 3) weight packs + BN fold (independent)
  k_pack<<<(9216 * 256) / 256, 256, 0, stream>>>(conv_w, wpc, 9216, 256, 0, 1024, 256);
  k_pack<<<(2304 * 256) / 256, 256, 0, stream>>>(mw1, wpm, 2304, 256, 0, 256, 256);
  k_pack<<<(256 * 256) / 256, 256, 0, stream>>>(cw1, wp1, 256, 256, 1, 0, 256);
  k_pack<<<(256 * 16) / 256, 256, 0, stream>>>(cw2, wp2, 256, 16, 1, 0, 6);
  k_bnfold<<<1, 256, 0, stream>>>(bn_g, bn_b, bn_m, bn_v, mb1, bnsc, bnbi);

  // 4) grid-sample into padded roi (reads featXcl)
  k_sample<<<8 * 1024, 256, 0, stream>>>(prop_ws, f0cl, f1cl, f2cl, f3cl, roipad);

  // 5) mask_feat -> padded CL bf16 (aliases featcl region; runs after sampling)
  k_tocl<<<8 * 128 * 8, 256, 0, stream>>>(mask_feat, mpad, 128, 128, (size_t)130 * 130 * 256,
                                          (size_t)130 * 256, (size_t)131 * 256);
  k_zero_ring<<<8 * 516, 256, 0, stream>>>(mpad, 130, 256);

  // 6) ROI conv GEMM (WMMA) -> bf16 channels-last
  k_gemm_roi<<<8 * 32 * 2, 32, 0, stream>>>(roipad, wpc, conv_b, roicl);

  // 7) classifier (WMMA), logits -> out[16384..65536)
  k_fc1<<<512, 32, 0, stream>>>(roicl, wp1, cb1, hbuf);
  k_fc2<<<512, 32, 0, stream>>>(hbuf, wp2, cb2, out + 16384);

  // 8) mask conv GEMM (WMMA) fused BN+ReLU+1x1
  k_gemm_mask<<<8 * 128 * 8, 32, 0, stream>>>(mpad, wpm, bnsc, bnbi, mw2, mb2, mdot);

  // 9) bilinear upsample -> out[65536..)
  k_upsample<<<(8 * 512 * 512) / 256, 256, 0, stream>>>(mdot, mb2, out + 65536);
}
